// MultiScaleRetention_2817498546686
// MI455X (gfx1250) — matrix operand order repaired and checked by reading, not compiled, not run on hardware
//
#include <hip/hip_runtime.h>

// ---------------------------------------------------------------------------
// MultiScaleRetention forward for MI455X (gfx1250).
// bf16 WMMA (f32 accum) for all matmuls; double-buffered TDM
// (tensor_load_to_lds) pipelining with s_wait_tensorcnt N for overlap.
// B=2, S=1024, HIDDEN=1024, HEADS=8, d=128.
// ---------------------------------------------------------------------------

typedef unsigned short     u16;
typedef unsigned int       u32;
typedef unsigned long long u64;

typedef __attribute__((ext_vector_type(16))) __bf16 v16bf;
typedef __attribute__((ext_vector_type(8)))  float  v8f;
typedef __attribute__((ext_vector_type(4)))  u32    v4u;
typedef __attribute__((ext_vector_type(8)))  int    v8i;
typedef __attribute__((ext_vector_type(4)))  int    v4i;

#define Bd   2
#define Sd   1024
#define Hd   8
#define Dd   128
#define HID  1024
#define NTOK (Bd * Sd)          // 2048 tokens

// ------------------------------ helpers ------------------------------------

__device__ __forceinline__ u16 f2bf(float x) {
  u32 u = __float_as_uint(x);
  u32 r = (u + 0x7FFFu + ((u >> 16) & 1u)) >> 16;
  return (u16)r;
}

union FragBF {
  v16bf v;
  u16   u[16];
  uint4 q[2];
};

__device__ __forceinline__ v8f vzero() {
  v8f z = {0.f, 0.f, 0.f, 0.f, 0.f, 0.f, 0.f, 0.f};
  return z;
}

// Generic pointer to a __shared__ object -> LDS byte offset (flat LDS
// aperture truncates to addr[31:0] per the ISA aperture mapping).
__device__ __forceinline__ u32 lds_off(const void* p) {
  return (u32)(uintptr_t)p;
}

// Tensor Data Mover: 2D bf16 tile (tile_w elems per row, tile_h rows,
// row stride stride_e elems) from global into LDS at lds_addr.
// D# per CDNA5 ISA ch.8: group0 = {flags, lds_addr, global_addr, type=2},
// group1 = {data_size=2B, tensor dims, tile dims, dim0 stride}.
// 6-arg builtin: (u32x4 g0, i32x8 g1, i32x4 g2, i32x4 g3, i32x8 aux, i32 cpol)
__device__ __forceinline__ void tdm_load_2d(u32 lds_addr, const u16* gptr,
                                            u32 tile_w, u32 tile_h,
                                            u32 stride_e) {
  u64 ga = (u64)(uintptr_t)gptr;
  v4u g0;
  g0[0] = 1u;                                        // count=1, user descriptor
  g0[1] = lds_addr;                                  // lds_addr [63:32]
  g0[2] = (u32)(ga & 0xFFFFFFFFu);                   // global_addr [95:64]
  g0[3] = (u32)((ga >> 32) & 0x01FFFFFFu) | 0x80000000u;  // addr[56:32] | type=2
  v8i g1;
  g1[0] = (int)(1u << 16);                           // data_size=1 (2 bytes)
  g1[1] = (int)((tile_w & 0xFFFFu) << 16);           // tensor_dim0[15:0] @63:48
  g1[2] = (int)(((tile_w >> 16) & 0xFFFFu) | ((tile_h & 0xFFFFu) << 16)); // d0 hi, d1 lo
  g1[3] = (int)(((tile_h >> 16) & 0xFFFFu) | ((tile_w & 0xFFFFu) << 16)); // d1 hi, tile_dim0
  g1[4] = (int)(tile_h & 0xFFFFu);                   // tile_dim1; tile_dim2=0
  g1[5] = (int)stride_e;                             // tensor_dim0_stride lo32
  g1[6] = 0;
  g1[7] = 0;
  v4i gz4 = {0, 0, 0, 0};
  v8i gz8 = {0, 0, 0, 0, 0, 0, 0, 0};
  __builtin_amdgcn_tensor_load_to_lds(g0, g1, gz4, gz4, gz8, 0);
}

// A-matrix fragment (16x32 bf16), row-major LDS tile, leading dim ld elems.
__device__ __forceinline__ v16bf load_frag_a(const u16* p, int ld) {
  int lane = threadIdx.x & 31;
  int m = lane & 15;
  int kh = lane >> 4;
  const u16* row = p + m * ld + kh * 8;
  FragBF f;
  f.q[0] = *(const uint4*)(row);
  f.q[1] = *(const uint4*)(row + 16);
  return f.v;
}

// B-matrix fragment (32x16 bf16), LDS stored n-major (row n = 32 consecutive k).
__device__ __forceinline__ v16bf load_frag_b(const u16* p, int ld) {
  int lane = threadIdx.x & 31;
  int n = lane & 15;
  int kh = lane >> 4;
  const u16* row = p + n * ld + kh * 8;
  FragBF f;
  f.q[0] = *(const uint4*)(row);
  f.q[1] = *(const uint4*)(row + 16);
  return f.v;
}

__device__ __forceinline__ v16bf neg_frag(v16bf a) {
  FragBF f;
  f.v = a;
#pragma unroll
  for (int i = 0; i < 16; ++i) f.u[i] ^= 0x8000u;
  return f.v;
}

__device__ __forceinline__ v8f wmma_bf16(v16bf a, v16bf b, v8f c) {
  return __builtin_amdgcn_wmma_f32_16x16x32_bf16(
      false, a, false, b, (short)0, c, false, false);
}

// ------------------------------ kernels ------------------------------------

__global__ void convert_bf16_k(const float* __restrict__ src,
                               u16* __restrict__ dst, int n) {
  int i = blockIdx.x * blockDim.x + threadIdx.x;
  if (i < n) dst[i] = f2bf(src[i]);
}

// nmat matrices of [rows][cols] f32 -> bf16 transposed [cols][rows].
__global__ void transpose_bf16_k(const float* __restrict__ src,
                                 u16* __restrict__ dst,
                                 int rows, int cols, int nmat) {
  int i = blockIdx.x * blockDim.x + threadIdx.x;
  int per = rows * cols;
  if (i >= per * nmat) return;
  int m = i / per, r = i - m * per;
  int n = r / rows, k = r - n * rows;
  dst[(size_t)m * per + (size_t)n * rows + k] =
      f2bf(src[(size_t)m * per + (size_t)k * cols + n]);
}

// QKV projection per head + rotation.  Grid (S/16, H, B), block 128 (4 waves).
// Weights pre-transposed Wt[h][n][k].  Double-buffered TDM (2 loads/tile).
__global__ __launch_bounds__(128) void qkv_rope_k(
    const u16* __restrict__ Xbf,
    const u16* __restrict__ Wqt, const u16* __restrict__ Wkt,
    const u16* __restrict__ Wvt, const float* __restrict__ theta,
    u16* __restrict__ Qc, u16* __restrict__ Qs,
    u16* __restrict__ Kc, u16* __restrict__ Ks, u16* __restrict__ Vb) {
  int tq = blockIdx.x, h = blockIdx.y, b = blockIdx.z;
  int s0 = tq * 16;
  int tid = threadIdx.x, lane = tid & 31, wid = tid >> 5;

  __shared__ __align__(16) u16 As[2][16 * 32];
  __shared__ __align__(16) u16 Bs[2][128 * 32];

  size_t headbase = (size_t)(b * Hd + h) * Sd * Dd;
  const u16* xrow = Xbf + (size_t)(b * Sd + s0) * HID + h * Dd;

  for (int w = 0; w < 3; ++w) {
    const u16* Wt = (w == 0) ? Wqt : ((w == 1) ? Wkt : Wvt);
    Wt += (size_t)h * Dd * Dd;
    v8f acc[2];
    acc[0] = vzero(); acc[1] = vzero();

    if (wid == 0) {                              // prologue: tile 0 -> buf 0
      tdm_load_2d(lds_off(As[0]), xrow, 32, 16, HID);
      tdm_load_2d(lds_off(Bs[0]), Wt, 32, 128, Dd);
    }
#pragma unroll
    for (int it = 0; it < 4; ++it) {
      int k0n = (it + 1) * 32;
      if (wid == 0) {
        if (k0n < Dd) {                          // issue tile it+1, keep it in flight
          tdm_load_2d(lds_off(As[(it + 1) & 1]), xrow + k0n, 32, 16, HID);
          tdm_load_2d(lds_off(Bs[(it + 1) & 1]), Wt + k0n, 32, 128, Dd);
          __builtin_amdgcn_s_wait_tensorcnt(2);  // tile it complete
        } else {
          __builtin_amdgcn_s_wait_tensorcnt(0);  // drain
        }
      }
      __syncthreads();
      int p = it & 1;
      v16bf a = load_frag_a(As[p], 32);
#pragma unroll
      for (int t = 0; t < 2; ++t) {
        v16bf bfr = load_frag_b(Bs[p] + (wid * 2 + t) * 16 * 32, 32);
        acc[t] = wmma_bf16(a, bfr, acc[t]);
      }
      __syncthreads();                           // reads done before overwrite
    }

#pragma unroll
    for (int t = 0; t < 2; ++t) {
      int ncol = (wid * 2 + t) * 16 + (lane & 15);
      int rbase = (lane >> 4) << 3;
#pragma unroll
      for (int v = 0; v < 8; ++v) {
        int srow = s0 + rbase + v;
        float val = acc[t][v];
        size_t o = headbase + (size_t)srow * Dd + ncol;
        if (w == 2) {
          Vb[o] = f2bf(val);
        } else {
          float th = theta[h * Dd + ncol];
          float ph = (float)(srow + 1) * th;   // ns = 1..S
          float c = __cosf(ph), s = __sinf(ph);
          if (w == 0) { Qc[o] = f2bf(val * c); Qs[o] = f2bf(val * s); }
          else        { Kc[o] = f2bf(val * c); Ks[o] = f2bf(val * s); }
        }
      }
    }
  }
}

// Retention attention.  Grid (S/16, H, B), block 32 (one wave / query tile).
// Double-buffered TDM on the 32-key K/V tiles (3 loads/tile).
__global__ __launch_bounds__(32) void retention_k(
    const u16* __restrict__ Qc, const u16* __restrict__ Qs,
    const u16* __restrict__ Kc, const u16* __restrict__ Ks,
    const u16* __restrict__ Vb,
    float* __restrict__ Yre, float* __restrict__ Yim) {
  int tq = blockIdx.x, h = blockIdx.y, b = blockIdx.z;
  int n0 = tq * 16;
  int lane = threadIdx.x;

  __shared__ __align__(16) u16 Qc_s[16 * 128];
  __shared__ __align__(16) u16 Qs_s[16 * 128];
  __shared__ __align__(16) u16 Kc_s[2][32 * 128];   // [key][d]
  __shared__ __align__(16) u16 Ks_s[2][32 * 128];
  __shared__ __align__(16) u16 Vn_s[2][32 * 128];   // [key][d] natural (TDM)
  __shared__ __align__(16) u16 Vt_s[128 * 32];      // [d][key] (n-major for B)
  __shared__ __align__(16) u16 Ar_s[16 * 32];
  __shared__ __align__(16) u16 Ai_s[16 * 32];

  size_t headbase = (size_t)(b * Hd + h) * Sd * Dd;
  const u16* kcg = Kc + headbase;
  const u16* ksg = Ks + headbase;
  const u16* vg  = Vb + headbase;

  // gamma_h = 1 - exp(linspace(log(1/32), log(1/512), 8))
  float l32 = __logf(1.f / 32.f), l512 = __logf(1.f / 512.f);
  float gm = 1.f - __expf(l32 + (l512 - l32) * ((float)h / 7.f));
  float lg = __logf(gm);

  v8f yre[8], yim[8];
#pragma unroll
  for (int t = 0; t < 8; ++t) { yre[t] = vzero(); yim[t] = vzero(); }

  // prologue: Q tiles + K/V tile 0 (in-order on TENSORcnt)
  tdm_load_2d(lds_off(Qc_s), Qc + headbase + (size_t)n0 * Dd, 128, 16, Dd);
  tdm_load_2d(lds_off(Qs_s), Qs + headbase + (size_t)n0 * Dd, 128, 16, Dd);
  tdm_load_2d(lds_off(Kc_s[0]), kcg, 128, 32, Dd);
  tdm_load_2d(lds_off(Ks_s[0]), ksg, 128, 32, Dd);
  tdm_load_2d(lds_off(Vn_s[0]), vg,  128, 32, Dd);

  int nsteps = (n0 + 16 + 31) / 32;
  for (int it = 0; it < nsteps; ++it) {
    int m0 = it * 32;   // all rows m0..m0+31 provably < S
    int pn = (it + 1) & 1;
    if (it + 1 < nsteps) {                       // issue next K/V tile
      size_t mn = (size_t)(m0 + 32) * Dd;
      tdm_load_2d(lds_off(Kc_s[pn]), kcg + mn, 128, 32, Dd);
      tdm_load_2d(lds_off(Ks_s[pn]), ksg + mn, 128, 32, Dd);
      tdm_load_2d(lds_off(Vn_s[pn]), vg  + mn, 128, 32, Dd);
      __builtin_amdgcn_s_wait_tensorcnt(3);      // tile it (+Q) complete
    } else {
      __builtin_amdgcn_s_wait_tensorcnt(0);      // drain
    }
    __syncthreads();
    int p = it & 1;
    for (int i = lane; i < 32 * 128; i += 32) {  // V transpose, LDS->LDS
      int r = i >> 7, dc = i & 127;
      Vt_s[dc * 32 + r] = Vn_s[p][r * 128 + dc];
    }
    __syncthreads();

#pragma unroll
    for (int hx = 0; hx < 2; ++hx) {
      v8f ar = vzero(), ai = vzero();
#pragma unroll
      for (int k0 = 0; k0 < Dd; k0 += 32) {
        v16bf aQc = load_frag_a(Qc_s + k0, 128);
        v16bf aQs = load_frag_a(Qs_s + k0, 128);
        v16bf bKc = load_frag_b(Kc_s[p] + hx * 16 * 128 + k0, 128);
        v16bf bKs = load_frag_b(Ks_s[p] + hx * 16 * 128 + k0, 128);
        ar = wmma_bf16(aQc, bKc, ar);
        ar = wmma_bf16(aQs, bKs, ar);
        ai = wmma_bf16(aQs, bKc, ai);
        ai = wmma_bf16(neg_frag(aQc), bKs, ai);   // - Qc.Ks^T
      }
      int mcol = m0 + hx * 16 + (lane & 15);
      int rbase = (lane >> 4) << 3;
#pragma unroll
      for (int v = 0; v < 8; ++v) {
        int n = n0 + rbase + v;
        float dec = (n >= mcol) ? __expf(lg * (float)(n - mcol)) : 0.f;
        int rl = rbase + v;
        Ar_s[rl * 32 + hx * 16 + (lane & 15)] = f2bf(ar[v] * dec);
        Ai_s[rl * 32 + hx * 16 + (lane & 15)] = f2bf(ai[v] * dec);
      }
    }
    __syncthreads();

    v16bf aR = load_frag_a(Ar_s, 32);
    v16bf aI = load_frag_a(Ai_s, 32);
#pragma unroll
    for (int ct = 0; ct < 8; ++ct) {
      v16bf bV = load_frag_b(Vt_s + ct * 16 * 32, 32);
      yre[ct] = wmma_bf16(aR, bV, yre[ct]);
      yim[ct] = wmma_bf16(aI, bV, yim[ct]);
    }
    __syncthreads();                             // reads done before overwrite
  }

  int rbase = (lane >> 4) << 3;
#pragma unroll
  for (int ct = 0; ct < 8; ++ct) {
#pragma unroll
    for (int v = 0; v < 8; ++v) {
      int srow = n0 + rbase + v;
      int col = h * Dd + ct * 16 + (lane & 15);
      size_t o = (size_t)(b * Sd + srow) * HID + col;
      Yre[o] = yre[ct][v];
      Yim[o] = yim[ct][v];
    }
  }
}

// Complex group norm, in place on Yre/Yim.  Grid NTOK*H, block 128.
__global__ __launch_bounds__(128) void group_norm_k(
    float* __restrict__ Yre, float* __restrict__ Yim,
    const float* __restrict__ w, const float* __restrict__ bias) {
  int bid = blockIdx.x;
  int token = bid >> 3, h = bid & 7;
  int tid = threadIdx.x;
  size_t o = (size_t)token * HID + h * Dd + tid;
  float re = Yre[o], im = Yim[o];

  __shared__ float sre[128], sim[128];
  sre[tid] = re; sim[tid] = im;
  __syncthreads();
  for (int s = 64; s > 0; s >>= 1) {
    if (tid < s) { sre[tid] += sre[tid + s]; sim[tid] += sim[tid + s]; }
    __syncthreads();
  }
  float mre = sre[0] * (1.f / 128.f);
  float mim = sim[0] * (1.f / 128.f);
  __syncthreads();
  float dre = re - mre, dim_ = im - mim;
  sre[tid] = dre * dre + dim_ * dim_;
  __syncthreads();
  for (int s = 64; s > 0; s >>= 1) {
    if (tid < s) sre[tid] += sre[tid + s];
    __syncthreads();
  }
  float var = sre[0] * (1.f / 127.f);       // unbiased, HEAD-1
  float inv = rsqrtf(var + 1e-5f);
  int col = h * Dd + tid;
  Yre[o] = (dre * inv) * w[col] + bias[col];
  Yim[o] = (dim_ * inv) * w[col];
}

// Gate path: G = X@Wg (complex), complex swish, Z = swish(G)+Y -> bf16.
// Grid (NTOK/16, HID/128), block 128.  Double-buffered TDM (3 loads/tile).
__global__ __launch_bounds__(128) void gate_combine_k(
    const u16* __restrict__ Xbf,
    const u16* __restrict__ Wgret, const u16* __restrict__ Wgimt,
    const float* __restrict__ Yre, const float* __restrict__ Yim,
    u16* __restrict__ Zre, u16* __restrict__ Zim) {
  int r0 = blockIdx.x * 16, cg = blockIdx.y;
  int tid = threadIdx.x, lane = tid & 31, wid = tid >> 5;

  __shared__ __align__(16) u16 As[2][16 * 32];
  __shared__ __align__(16) u16 Br[2][128 * 32];
  __shared__ __align__(16) u16 Bi[2][128 * 32];

  v8f gre[2], gim[2];
  gre[0] = vzero(); gre[1] = vzero(); gim[0] = vzero(); gim[1] = vzero();

  const u16* xr = Xbf + (size_t)r0 * HID;
  const u16* wr = Wgret + (size_t)cg * 128 * HID;
  const u16* wi = Wgimt + (size_t)cg * 128 * HID;

  if (wid == 0) {                                // prologue: tile 0
    tdm_load_2d(lds_off(As[0]), xr, 32, 16, HID);
    tdm_load_2d(lds_off(Br[0]), wr, 32, 128, HID);
    tdm_load_2d(lds_off(Bi[0]), wi, 32, 128, HID);
  }
  for (int it = 0; it < HID / 32; ++it) {
    int k0n = (it + 1) * 32;
    if (wid == 0) {
      if (k0n < HID) {
        int pn = (it + 1) & 1;
        tdm_load_2d(lds_off(As[pn]), xr + k0n, 32, 16, HID);
        tdm_load_2d(lds_off(Br[pn]), wr + k0n, 32, 128, HID);
        tdm_load_2d(lds_off(Bi[pn]), wi + k0n, 32, 128, HID);
        __builtin_amdgcn_s_wait_tensorcnt(3);    // tile it complete
      } else {
        __builtin_amdgcn_s_wait_tensorcnt(0);    // drain
      }
    }
    __syncthreads();
    int p = it & 1;
    v16bf a = load_frag_a(As[p], 32);
#pragma unroll
    for (int t = 0; t < 2; ++t) {
      v16bf br = load_frag_b(Br[p] + (wid * 2 + t) * 16 * 32, 32);
      v16bf bi = load_frag_b(Bi[p] + (wid * 2 + t) * 16 * 32, 32);
      gre[t] = wmma_bf16(a, br, gre[t]);
      gim[t] = wmma_bf16(a, bi, gim[t]);
    }
    __syncthreads();
  }

  int rbase = (lane >> 4) << 3;
#pragma unroll
  for (int t = 0; t < 2; ++t) {
    int col = cg * 128 + (wid * 2 + t) * 16 + (lane & 15);
#pragma unroll
    for (int v = 0; v < 8; ++v) {
      int token = r0 + rbase + v;
      float a = gre[t][v], bb = gim[t][v];
      // s = 1 / (1 + e^{-a}(cos b - i sin b)), G_sw = G*s
      float e = __expf(-a);
      float dr = 1.f + e * __cosf(bb);
      float di = -e * __sinf(bb);
      float den = dr * dr + di * di;
      float sr = dr / den, si = -di / den;
      float gr = a * sr - bb * si;
      float gi = a * si + bb * sr;
      size_t o = (size_t)token * HID + col;
      Zre[o] = f2bf(gr + Yre[o]);
      Zim[o] = f2bf(gi + Yim[o]);
    }
  }
}

// Final complex projection: out = Z @ (Wo_re + i Wo_im), interleaved f32 out.
// Double-buffered TDM (4 loads/tile).
__global__ __launch_bounds__(128) void final_out_k(
    const u16* __restrict__ Zre, const u16* __restrict__ Zim,
    const u16* __restrict__ Woret, const u16* __restrict__ Woimt,
    float* __restrict__ Out) {
  int r0 = blockIdx.x * 16, cg = blockIdx.y;
  int tid = threadIdx.x, lane = tid & 31, wid = tid >> 5;

  __shared__ __align__(16) u16 Ar[2][16 * 32];
  __shared__ __align__(16) u16 Ai[2][16 * 32];
  __shared__ __align__(16) u16 Br[2][128 * 32];
  __shared__ __align__(16) u16 Bi[2][128 * 32];

  v8f ore[2], oim[2];
  ore[0] = vzero(); ore[1] = vzero(); oim[0] = vzero(); oim[1] = vzero();

  const u16* zr = Zre + (size_t)r0 * HID;
  const u16* zi = Zim + (size_t)r0 * HID;
  const u16* wr = Woret + (size_t)cg * 128 * HID;
  const u16* wi = Woimt + (size_t)cg * 128 * HID;

  if (wid == 0) {                                // prologue: tile 0
    tdm_load_2d(lds_off(Ar[0]), zr, 32, 16, HID);
    tdm_load_2d(lds_off(Ai[0]), zi, 32, 16, HID);
    tdm_load_2d(lds_off(Br[0]), wr, 32, 128, HID);
    tdm_load_2d(lds_off(Bi[0]), wi, 32, 128, HID);
  }
  for (int it = 0; it < HID / 32; ++it) {
    int k0n = (it + 1) * 32;
    if (wid == 0) {
      if (k0n < HID) {
        int pn = (it + 1) & 1;
        tdm_load_2d(lds_off(Ar[pn]), zr + k0n, 32, 16, HID);
        tdm_load_2d(lds_off(Ai[pn]), zi + k0n, 32, 16, HID);
        tdm_load_2d(lds_off(Br[pn]), wr + k0n, 32, 128, HID);
        tdm_load_2d(lds_off(Bi[pn]), wi + k0n, 32, 128, HID);
        __builtin_amdgcn_s_wait_tensorcnt(4);    // tile it complete
      } else {
        __builtin_amdgcn_s_wait_tensorcnt(0);    // drain
      }
    }
    __syncthreads();
    int p = it & 1;
    v16bf ar = load_frag_a(Ar[p], 32);
    v16bf ai = load_frag_a(Ai[p], 32);
    v16bf nai = neg_frag(ai);
#pragma unroll
    for (int t = 0; t < 2; ++t) {
      v16bf br = load_frag_b(Br[p] + (wid * 2 + t) * 16 * 32, 32);
      v16bf bi = load_frag_b(Bi[p] + (wid * 2 + t) * 16 * 32, 32);
      ore[t] = wmma_bf16(ar, br, ore[t]);
      ore[t] = wmma_bf16(nai, bi, ore[t]);   // re = Zr.Wor - Zi.Woi
      oim[t] = wmma_bf16(ar, bi, oim[t]);
      oim[t] = wmma_bf16(ai, br, oim[t]);    // im = Zr.Woi + Zi.Wor
    }
    __syncthreads();
  }

  int rbase = (lane >> 4) << 3;
#pragma unroll
  for (int t = 0; t < 2; ++t) {
    int col = cg * 128 + (wid * 2 + t) * 16 + (lane & 15);
#pragma unroll
    for (int v = 0; v < 8; ++v) {
      int token = r0 + rbase + v;
      size_t o = (size_t)token * HID + col;
      Out[o * 2 + 0] = ore[t][v];   // complex64 interleaved
      Out[o * 2 + 1] = oim[t][v];
    }
  }
}

// ------------------------------ launch -------------------------------------

extern "C" void kernel_launch(void* const* d_in, const int* in_sizes, int n_in,
                              void* d_out, int out_size, void* d_ws,
                              size_t ws_size, hipStream_t stream) {
  (void)in_sizes; (void)n_in; (void)out_size; (void)ws_size;

  const float* X     = (const float*)d_in[0];
  const float* Wq    = (const float*)d_in[1];
  const float* Wk    = (const float*)d_in[2];
  const float* Wv    = (const float*)d_in[3];
  const float* theta = (const float*)d_in[4];
  const float* gnw   = (const float*)d_in[5];
  const float* gnb   = (const float*)d_in[6];
  const float* Wgre  = (const float*)d_in[7];
  const float* Wgim  = (const float*)d_in[8];
  const float* Wore  = (const float*)d_in[9];
  const float* Woim  = (const float*)d_in[10];
  float* Out = (float*)d_out;

  constexpr size_t N_X  = (size_t)NTOK * HID;
  constexpr size_t N_WH = (size_t)Hd * Dd * Dd;
  constexpr size_t N_WB = (size_t)HID * HID;
  constexpr size_t N_PH = (size_t)Bd * Hd * Sd * Dd;

  char* ws = (char*)d_ws;
  size_t off = 0;
  auto take = [&](size_t bytes) {
    char* p = ws + off;
    off += (bytes + 255) & ~(size_t)255;
    return p;
  };

  u16* xbf   = (u16*)take(N_X * 2);
  u16* wq_t  = (u16*)take(N_WH * 2);
  u16* wk_t  = (u16*)take(N_WH * 2);
  u16* wv_t  = (u16*)take(N_WH * 2);
  u16* wgr_t = (u16*)take(N_WB * 2);
  u16* wgi_t = (u16*)take(N_WB * 2);
  u16* wor_t = (u16*)take(N_WB * 2);
  u16* woi_t = (u16*)take(N_WB * 2);
  u16* qc    = (u16*)take(N_PH * 2);
  u16* qs    = (u16*)take(N_PH * 2);
  u16* kc    = (u16*)take(N_PH * 2);
  u16* ks    = (u16*)take(N_PH * 2);
  u16* vb    = (u16*)take(N_PH * 2);
  float* yre = (float*)take(N_X * 4);
  float* yim = (float*)take(N_X * 4);
  u16* zre   = (u16*)take(N_X * 2);
  u16* zim   = (u16*)take(N_X * 2);

  convert_bf16_k<<<dim3((unsigned)((N_X + 255) / 256)), dim3(256), 0, stream>>>(
      X, xbf, (int)N_X);

  auto transp = [&](const float* s, u16* d, int rows, int cols, int nmat) {
    size_t n = (size_t)rows * cols * nmat;
    transpose_bf16_k<<<dim3((unsigned)((n + 255) / 256)), dim3(256), 0, stream>>>(
        s, d, rows, cols, nmat);
  };
  transp(Wq, wq_t, Dd, Dd, Hd);
  transp(Wk, wk_t, Dd, Dd, Hd);
  transp(Wv, wv_t, Dd, Dd, Hd);
  transp(Wgre, wgr_t, HID, HID, 1);
  transp(Wgim, wgi_t, HID, HID, 1);
  transp(Wore, wor_t, HID, HID, 1);
  transp(Woim, woi_t, HID, HID, 1);

  qkv_rope_k<<<dim3(Sd / 16, Hd, Bd), dim3(128), 0, stream>>>(
      xbf, wq_t, wk_t, wv_t, theta, qc, qs, kc, ks, vb);

  retention_k<<<dim3(Sd / 16, Hd, Bd), dim3(32), 0, stream>>>(
      qc, qs, kc, ks, vb, yre, yim);

  group_norm_k<<<dim3(NTOK * Hd), dim3(128), 0, stream>>>(yre, yim, gnw, gnb);

  gate_combine_k<<<dim3(NTOK / 16, HID / 128), dim3(128), 0, stream>>>(
      xbf, wgr_t, wgi_t, yre, yim, zre, zim);

  final_out_k<<<dim3(NTOK / 16, HID / 128), dim3(128), 0, stream>>>(
      zre, zim, wor_t, woi_t, Out);
}
